// AttModel_16097537426007
// MI455X (gfx1250) — matrix operand order repaired and checked by reading, not compiled, not run on hardware
//
#include <hip/hip_runtime.h>
#include <hip/hip_bf16.h>

typedef __attribute__((ext_vector_type(16))) _Float16 v16h;
typedef __attribute__((ext_vector_type(4)))  _Float16 v4h;
typedef __attribute__((ext_vector_type(8)))  float    v8f;

#define DIN  128
#define HID  128
#define DOUT 128
#define NN   64     // N == attention K dim

__device__ __forceinline__ v8f wmma16(v16h a, v16h b, v8f c) {
    return __builtin_amdgcn_wmma_f32_16x16x32_f16(false, a, false, b,
                                                  (short)0, c, false, false);
}

// A matrix 16x32 (MxK) f16 from LDS, row-major, leading dim ld.
// Lane L holds row m=L&15; K runs [hi*16, hi*16+16) contiguous
// (matches B fragment k order).
__device__ __forceinline__ v16h frag_A_lds(const _Float16* base, int ld, int lane) {
    const int m  = lane & 15;
    const int hi = lane >> 4;
    const _Float16* row = base + m * ld;
    v16h a;
#pragma unroll
    for (int v = 0; v < 8; ++v) {
        const int kb = (v << 1) + (hi << 4);
        a[2 * v]     = row[kb];
        a[2 * v + 1] = row[kb + 1];
    }
    return a;
}

// B matrix 32x16 (KxN) fragment from a *transposed* f16 source T[n][k]
// (leading dim ldk): per lane col n=L&15, K=hi*16+e contiguous => one v16h.
__device__ __forceinline__ v16h frag_B_T(const _Float16* tbase, int ldk, int lane) {
    const int n  = lane & 15;
    const int hi = lane >> 4;
    return *(const v16h*)(tbase + n * ldk + (hi << 4));
}

// Fallback: B fragment gathered from row-major f32 global weights.
__device__ __forceinline__ v16h frag_B_glb(const float* base, int ld, int lane) {
    const int n  = lane & 15;
    const int hi = lane >> 4;
    v16h b;
#pragma unroll
    for (int e = 0; e < 16; ++e)
        b[e] = (_Float16)base[(e + (hi << 4)) * ld + n];
    return b;
}

// ---- prep: Wk/Wv -> transposed f16 copies in workspace ---------------------
__global__ __launch_bounds__(256) void prep_weights(
    const float* __restrict__ Wk, const float* __restrict__ Wv,
    _Float16* __restrict__ WkT, _Float16* __restrict__ WvT)
{
    const int idx = blockIdx.x * 256 + threadIdx.x;   // 16384 total
    const int d = idx >> 7;
    const int h = idx & 127;
    WkT[h * HID + d] = (_Float16)Wk[d * HID + h];
    WvT[h * HID + d] = (_Float16)Wv[d * HID + h];
}

// ---- Fully fused attention model: one block (8 wave32) per (b, n) ---------
template <bool USE_WS>
__global__ __launch_bounds__(256) void attmodel_fused(
    const float* __restrict__ obs, const float* __restrict__ adj,
    const float* __restrict__ Wq,  const float* __restrict__ bq,
    const float* __restrict__ Wk,  const float* __restrict__ bk,
    const float* __restrict__ Wv,  const float* __restrict__ bv,
    const float* __restrict__ Wo,  const float* __restrict__ bo,
    const _Float16* __restrict__ WkT, const _Float16* __restrict__ WvT,
    float* __restrict__ out)
{
    __shared__ _Float16 s_obsT[DIN * NN];  // 16 KB  obs[b]^T  : [d][m]
    __shared__ _Float16 s_adj [NN * NN];   //  8 KB  adj[b,n]  : [k][m]
    __shared__ _Float16 s_hc  [NN * HID];  // 16 KB  hc        : [k][d]
    __shared__ float    s_q[HID];
    __shared__ float    s_logits[NN];      // raw logits, later normalized att
    __shared__ float    s_att[NN];         // exp scratch
    __shared__ float    s_agg[HID];

    const int tid  = threadIdx.x;
    const int lane = tid & 31;             // wave32
    const int w    = tid >> 5;             // wave id 0..7 -> owns cols [w*16, w*16+16)
    const int bid  = blockIdx.x;           // = b*64 + n
    const int b    = bid >> 6;
    const int n    = bid & 63;

    const int col = lane & 15;
    const int hi  = lane >> 4;
    const int h   = w * 16 + col;          // this lane's output-column index

    const float* obs_b = obs + (size_t)b * NN * DIN;
    const float* adj_t = adj + (size_t)bid * NN * NN;

    // Prefetch next block's adj tile (streaming 268 MB from HBM).
    if (bid + 1 < 256 * 64)
        __builtin_prefetch(adj + (size_t)(bid + 1) * NN * NN + tid * 16, 0, 0);

    // ---- Phase 0: stage adj (row-major f16) + obs (transposed f16) ----
    {
        const float4* a4 = (const float4*)adj_t;
        for (int i = tid; i < NN * NN / 4; i += 256) {
            float4 v = a4[i];
            v4h p; p[0] = (_Float16)v.x; p[1] = (_Float16)v.y;
                   p[2] = (_Float16)v.z; p[3] = (_Float16)v.w;
            *(v4h*)(s_adj + i * 4) = p;            // ds_store_b64
        }
        const float4* o4 = (const float4*)obs_b;
        for (int i = tid; i < NN * DIN / 4; i += 256) {
            float4 v = o4[i];
            const int m  = (i * 4) >> 7;
            const int d0 = (i * 4) & 127;
            s_obsT[(d0 + 0) * NN + m] = (_Float16)v.x;
            s_obsT[(d0 + 1) * NN + m] = (_Float16)v.y;
            s_obsT[(d0 + 2) * NN + m] = (_Float16)v.z;
            s_obsT[(d0 + 3) * NN + m] = (_Float16)v.w;
        }
    }
    if (tid < NN)  s_logits[tid] = 0.0f;
    if (tid < HID) s_agg[tid]    = 0.0f;
    __syncthreads();

    // ---- Phase 1a: q = relu(obs[b,n] @ Wq + bq) (LDS broadcast reads) ----
    if (tid < HID) {
        float acc = bq[tid];
#pragma unroll 8
        for (int d = 0; d < DIN; ++d)
            acc += (float)s_obsT[d * NN + n] * Wq[d * HID + tid];
        s_q[tid] = fmaxf(acc, 0.0f);
    }

    // ---- Phase 1b: hc = adj @ obs (M=64,N=128,K=64) -> s_hc f16 ----
    {
        v16h bfr[2];
#pragma unroll
        for (int kt = 0; kt < 2; ++kt)   // preload both B fragments
            bfr[kt] = frag_B_T(s_obsT + (w * 16) * NN + kt * 32, NN, lane);
        v8f acc[4] = {};
#pragma unroll
        for (int kt = 0; kt < 2; ++kt)
#pragma unroll
            for (int mt = 0; mt < 4; ++mt) {
                v16h a = frag_A_lds(s_adj + mt * 16 * NN + kt * 32, NN, lane);
                acc[mt] = wmma16(a, bfr[kt], acc[mt]);
            }
#pragma unroll
        for (int mt = 0; mt < 4; ++mt)
#pragma unroll
            for (int v = 0; v < 8; ++v)
                s_hc[(mt * 16 + v + hi * 8) * HID + w * 16 + col] = (_Float16)acc[mt][v];
    }
    __syncthreads();

    // ---- Phase 2: key = relu(hc @ Wk + bk); logits[k] += q . key[k,:] ----
    {
        const float qh  = s_q[h];
        const float bkh = bk[h];
        v16h bfr[4];
#pragma unroll
        for (int kt = 0; kt < 4; ++kt)   // preload all weight fragments (L2)
            bfr[kt] = USE_WS
                ? frag_B_T(WkT + (w * 16) * HID + kt * 32, HID, lane)
                : frag_B_glb(Wk + (kt * 32) * HID + w * 16, HID, lane);
        v8f acc[4] = {};
#pragma unroll
        for (int kt = 0; kt < 4; ++kt)
#pragma unroll
            for (int mt = 0; mt < 4; ++mt) {
                v16h a = frag_A_lds(s_hc + mt * 16 * HID + kt * 32, HID, lane);
                acc[mt] = wmma16(a, bfr[kt], acc[mt]);
            }
#pragma unroll
        for (int mt = 0; mt < 4; ++mt) {
#pragma unroll
            for (int v = 0; v < 8; ++v) {
                float kv = fmaxf(acc[mt][v] + bkh, 0.0f) * qh;
#pragma unroll
                for (int off = 8; off; off >>= 1)      // reduce over 16 cols
                    kv += __shfl_xor(kv, off, 32);
                if (col == 0)
                    atomicAdd(&s_logits[mt * 16 + v + hi * 8], kv);
            }
        }
    }
    __syncthreads();

    // ---- Phase 3: softmax over the 64 logits ----
    if (tid < NN) {
        float mx = -1e30f;
#pragma unroll
        for (int k = 0; k < NN; ++k) mx = fmaxf(mx, s_logits[k]);
        s_att[tid] = __expf(s_logits[tid] - mx);
    }
    __syncthreads();
    if (tid < NN) {
        float ssum = 0.0f;
#pragma unroll
        for (int k = 0; k < NN; ++k) ssum += s_att[k];
        s_logits[tid] = s_att[tid] / ssum;   // s_logits now holds att
    }
    __syncthreads();

    // ---- Phase 4: value = relu(hc @ Wv + bv); agg[h] += sum_k att[k]*value ----
    {
        const float bvh = bv[h];
        v16h bfr[4];
#pragma unroll
        for (int kt = 0; kt < 4; ++kt)   // preload all weight fragments (L2)
            bfr[kt] = USE_WS
                ? frag_B_T(WvT + (w * 16) * HID + kt * 32, HID, lane)
                : frag_B_glb(Wv + (kt * 32) * HID + w * 16, HID, lane);
        v8f acc[4] = {};
#pragma unroll
        for (int kt = 0; kt < 4; ++kt)
#pragma unroll
            for (int mt = 0; mt < 4; ++mt) {
                v16h a = frag_A_lds(s_hc + mt * 16 * HID + kt * 32, HID, lane);
                acc[mt] = wmma16(a, bfr[kt], acc[mt]);
            }
        float part = 0.0f;
#pragma unroll
        for (int mt = 0; mt < 4; ++mt)
#pragma unroll
            for (int v = 0; v < 8; ++v)
                part += s_logits[mt * 16 + v + hi * 8] *
                        fmaxf(acc[mt][v] + bvh, 0.0f);
        atomicAdd(&s_agg[h], part);   // 2 lanes per h (hi=0/1)
    }
    __syncthreads();

    // ---- Phase 5: out = relu(agg @ Wo + bo) ----
    if (tid < DOUT) {
        float acc = bo[tid];
#pragma unroll 8
        for (int k = 0; k < HID; ++k)
            acc += s_agg[k] * Wo[k * DOUT + tid];
        out[(size_t)bid * DOUT + tid] = fmaxf(acc, 0.0f);
    }
}

extern "C" void kernel_launch(void* const* d_in, const int* in_sizes, int n_in,
                              void* d_out, int out_size, void* d_ws, size_t ws_size,
                              hipStream_t stream) {
    (void)in_sizes; (void)n_in; (void)out_size;
    const float* obs = (const float*)d_in[0];
    const float* adj = (const float*)d_in[1];
    const float* Wq  = (const float*)d_in[2];
    const float* bq  = (const float*)d_in[3];
    const float* Wk  = (const float*)d_in[4];
    const float* bk  = (const float*)d_in[5];
    const float* Wv  = (const float*)d_in[6];
    const float* bv  = (const float*)d_in[7];
    const float* Wo  = (const float*)d_in[8];
    const float* bo  = (const float*)d_in[9];
    float* out = (float*)d_out;

    const size_t wbytes = (size_t)HID * HID * sizeof(_Float16);   // 32 KB each
    const bool   use_ws = (d_ws != nullptr) && (ws_size >= 2 * wbytes);

    dim3 grid(256 * 64);   // one block per (b, n)
    dim3 block(256);       // 8 wave32

    if (use_ws) {
        _Float16* WkT = (_Float16*)d_ws;
        _Float16* WvT = WkT + (size_t)HID * HID;
        hipLaunchKernelGGL(prep_weights, dim3(HID * HID / 256), block, 0, stream,
                           Wk, Wv, WkT, WvT);
        hipLaunchKernelGGL((attmodel_fused<true>), grid, block, 0, stream,
                           obs, adj, Wq, bq, Wk, bk, Wv, bv, Wo, bo,
                           (const _Float16*)WkT, (const _Float16*)WvT, out);
    } else {
        hipLaunchKernelGGL((attmodel_fused<false>), grid, block, 0, stream,
                           obs, adj, Wq, bq, Wk, bk, Wv, bv, Wo, bo,
                           (const _Float16*)nullptr, (const _Float16*)nullptr, out);
    }
}
